// RBN_41798621724808
// MI455X (gfx1250) — matrix operand order repaired and checked by reading, not compiled, not run on hardware
//
#include <hip/hip_runtime.h>
#include <math.h>

// Problem constants (match reference)
constexpr int Bb  = 8;
constexpr int Cc  = 2048;
constexpr int Dd  = 64;
constexpr int Ll  = 3;
constexpr int Hh  = 4;
constexpr int DHd = 16;
constexpr int OUTn = 10;

typedef _Float16 v16h __attribute__((ext_vector_type(16)));
typedef float    v8f  __attribute__((ext_vector_type(8)));

#ifndef __has_builtin
#define __has_builtin(x) 0
#endif

#if __has_builtin(__builtin_amdgcn_global_load_async_to_lds_b32)
#define HAVE_ASYNC_LDS 1
typedef __attribute__((address_space(1))) int* gint_p;
typedef __attribute__((address_space(3))) int* lint_p;
#else
#define HAVE_ASYNC_LDS 0
#endif

__device__ __forceinline__ float gelu_f(float v) {
    return 0.5f * v * (1.0f + erff(v * 0.70710678118654752f));
}

__device__ __forceinline__ void wait_async_lds() {
#if __has_builtin(__builtin_amdgcn_s_wait_asynccnt)
    __builtin_amdgcn_s_wait_asynccnt(0);
#else
    asm volatile("s_wait_asynccnt 0x0" ::: "memory");
#endif
}

// ---------------------------------------------------------------------------
// h[b,c,d] = role_emb[c,d] * gelu(log1p(max(x,0)) * filler_w[d])
// ---------------------------------------------------------------------------
__global__ void bind_kernel(const float* __restrict__ x,
                            const float* __restrict__ role,
                            const float* __restrict__ fw,
                            float* __restrict__ h) {
    int idx = blockIdx.x * blockDim.x + threadIdx.x;
    if (idx >= Bb * Cc * Dd) return;
    int dd = idx & 63;
    int c  = (idx >> 6) & (Cc - 1);
    int b  = idx >> 17;
    float xl = log1pf(fmaxf(x[b * Cc + c], 0.0f));
    h[idx] = role[c * Dd + dd] * gelu_f(xl * fw[dd]);
}

// ---------------------------------------------------------------------------
// Per token: q/k/v = h @ W^T + b ; qp = q @ w1q^T ; kp = k @ w1k^T
// one block per token, 64 threads
// ---------------------------------------------------------------------------
__global__ void qkv_kernel(const float* __restrict__ h,
                           const float* __restrict__ qw, const float* __restrict__ qb,
                           const float* __restrict__ kw, const float* __restrict__ kb,
                           const float* __restrict__ vw, const float* __restrict__ vb,
                           const float* __restrict__ w1,   // [16,48] this layer
                           float* __restrict__ q, float* __restrict__ k,
                           float* __restrict__ v, float* __restrict__ qp,
                           float* __restrict__ kp) {
    __shared__ float sH[64], sQ[64], sK[64];
    const int tok = blockIdx.x;
    const int t   = threadIdx.x;
    sH[t] = h[(size_t)tok * Dd + t];
    __syncthreads();
    float aq = qb[t], ak = kb[t], av = vb[t];
    for (int dd = 0; dd < 64; ++dd) {
        float hv = sH[dd];
        aq += hv * qw[t * 64 + dd];
        ak += hv * kw[t * 64 + dd];
        av += hv * vw[t * 64 + dd];
    }
    q[(size_t)tok * Dd + t] = aq;
    k[(size_t)tok * Dd + t] = ak;
    v[(size_t)tok * Dd + t] = av;
    sQ[t] = aq; sK[t] = ak;
    __syncthreads();
    const int hd = t >> 4, e = t & 15;
    float pq = 0.f, pk = 0.f;
    for (int d = 0; d < 16; ++d) {
        pq += sQ[hd * 16 + d] * w1[e * 48 + d];
        pk += sK[hd * 16 + d] * w1[e * 48 + 16 + d];
    }
    qp[(size_t)tok * Dd + t] = pq;
    kp[(size_t)tok * Dd + t] = pk;
}

// ---------------------------------------------------------------------------
// Flash-attention with second-order MLP scoring via WMMA f16 (f32 accumulate).
// grid = (NC=32, H=4, B=8), block = 128 (4 waves, 16 queries each).
// Per key j: res[16q,16e] = WMMA( A=[q||qp] , B=[k*Wqk || I16] , C=kp+b1 )
// score = gelu(res)@w2 + b2, online softmax, P@V via a second WMMA (K=32 keys).
// k/v staging uses GLOBAL_LOAD_ASYNC_TO_LDS (ASYNCcnt) when available.
// ---------------------------------------------------------------------------
__global__ void __launch_bounds__(128)
attn_kernel(const float* __restrict__ q,  const float* __restrict__ qp,
            const float* __restrict__ k,  const float* __restrict__ kp,
            const float* __restrict__ v,
            const float* __restrict__ w1, const float* __restrict__ b1,
            const float* __restrict__ w2, const float* __restrict__ b2,
            float* __restrict__ attnOut) {
    __shared__ float sKf[32 * 16];
    __shared__ float sVf[32 * 16];
    __shared__ float sKPB[32 * 16];
    __shared__ float sScore[4 * 16 * 32];   // per wave: [16 q][32 j]

    const int chunk = blockIdx.x, head = blockIdx.y, b = blockIdx.z;
    const int tid = threadIdx.x, wave = tid >> 5, lane = tid & 31;
    const int hi = lane >> 4, row = lane & 15, base = hi * 8;
    const int qbase = chunk * 64 + wave * 16;
    const size_t bc = (size_t)b * Cc;

    // ---- A fragment: K 0..15 = q, K 16..31 = qp (identity-folded bias term)
    const float* qrow  = q  + (bc + qbase + row) * Dd + head * 16;
    const float* qprow = qp + (bc + qbase + row) * Dd + head * 16;
    v16h af;
#pragma unroll
    for (int t2 = 0; t2 < 16; ++t2) {
        af[t2] = (_Float16)(hi ? qprow[t2] : qrow[t2]);   // K = 16*hi+t2
    }

    // ---- per-lane constants for B fragment
    // lanes 0-15 hold K=0..15 (k_j[d]*Wqk[e,d]); lanes 16-31 hold K=16..31 (I16)
    float wA[16];
    v16h idf;
#pragma unroll
    for (int t2 = 0; t2 < 16; ++t2) {
        wA[t2]  = w1[row * 48 + 32 + t2];                       // Wqk[e=row, d=t2]
        idf[t2] = (_Float16)((t2 == row) ? 1.0f : 0.0f);        // identity block
    }
    const float w2v   = w2[row];
    const float b2v   = b2[0];
    const float scale = 0.25f;                                   // 1/sqrt(16)

    v8f oAcc;
#pragma unroll
    for (int r = 0; r < 8; ++r) oAcc[r] = 0.0f;
    float mRow = -INFINITY, ssum = 0.0f;
    float* sSc = sScore + wave * 16 * 32;

    for (int jb = 0; jb < Cc; jb += 32) {
        __syncthreads();
        // ---- stage k, v (async memory->LDS when available), kp+b1 (VALU add)
#pragma unroll
        for (int it = 0; it < 4; ++it) {
            int idx2 = tid * 4 + it;             // 0..511
            int j = idx2 >> 4, e = idx2 & 15;
            size_t g = (bc + jb + j) * Dd + head * 16 + e;
#if HAVE_ASYNC_LDS
            __builtin_amdgcn_global_load_async_to_lds_b32(
                (gint_p)(k + g), (lint_p)&sKf[idx2], 0, 0);
            __builtin_amdgcn_global_load_async_to_lds_b32(
                (gint_p)(v + g), (lint_p)&sVf[idx2], 0, 0);
#else
            sKf[idx2] = k[g];
            sVf[idx2] = v[g];
#endif
            sKPB[idx2] = kp[g] + b1[e];
        }
        if (jb + 32 < Cc) {                      // prefetch next key block
            size_t g = (bc + jb + 32 + (tid >> 2)) * Dd + head * 16;
            __builtin_prefetch(k + g, 0, 1);
            __builtin_prefetch(v + g, 0, 1);
        }
#if HAVE_ASYNC_LDS
        wait_async_lds();
#endif
        __syncthreads();

        // ---- scoring: one WMMA per key j (unroll 2: overlap WMMA with gelu)
#pragma unroll 2
        for (int j = 0; j < 32; ++j) {
            v16h bf;
            if (hi == 0) {
#pragma unroll
                for (int t2 = 0; t2 < 16; ++t2)
                    bf[t2] = (_Float16)(sKf[j * 16 + t2] * wA[t2]);
            } else {
                bf = idf;
            }
            float cv = sKPB[j * 16 + row];       // kp_j[e] + b1[e]
            v8f cin;
#pragma unroll
            for (int r = 0; r < 8; ++r) cin[r] = cv;
            v8f res = __builtin_amdgcn_wmma_f32_16x16x32_f16(
                false, af, false, bf, (short)0, cin, false, false);
            // score[m,j] = (sum_e gelu(res[m,e]) * w2[e] + b2) * scale
#pragma unroll
            for (int r = 0; r < 8; ++r) {
                float t3 = gelu_f(res[r]) * w2v;
                t3 += __shfl_xor(t3, 1, 32);
                t3 += __shfl_xor(t3, 2, 32);
                t3 += __shfl_xor(t3, 4, 32);
                t3 += __shfl_xor(t3, 8, 32);
                if (row == 0) sSc[(r + 8 * hi) * 32 + j] = (t3 + b2v) * scale;
            }
        }
        __syncthreads();

        // ---- online softmax for row `row` (lanes L and L+16 duplicate)
        float mb = -INFINITY;
        for (int j = 0; j < 32; ++j) mb = fmaxf(mb, sSc[row * 32 + j]);
        float mNew = fmaxf(mRow, mb);
        float corr = expf(mRow - mNew);
        float bsum = 0.0f;
        for (int j = 0; j < 32; ++j) bsum += expf(sSc[row * 32 + j] - mNew);
        ssum = ssum * corr + bsum;
        mRow = mNew;

        // P fragment (A layout): p[row, K(keys)]
        v16h pf, vf;
#pragma unroll
        for (int t2 = 0; t2 < 8; ++t2) {
            pf[t2]     = (_Float16)expf(sSc[row * 32 + base + t2]      - mNew);
            pf[8 + t2] = (_Float16)expf(sSc[row * 32 + 16 + base + t2] - mNew);
        }
        // V fragment (B layout): same key order as P (base+t2, 16+base+t2-8)
#pragma unroll
        for (int t2 = 0; t2 < 16; ++t2)
            vf[t2] = (_Float16)sVf[((t2 < 8 ? 0 : 8) + base + t2) * 16 + row];

        // rescale running output by per-row correction, then accumulate P@V
#pragma unroll
        for (int r = 0; r < 8; ++r) oAcc[r] *= __shfl(corr, r + 8 * hi, 32);
        oAcc = __builtin_amdgcn_wmma_f32_16x16x32_f16(
            false, pf, false, vf, (short)0, oAcc, false, false);
    }

    // ---- finalize: divide by softmax denominator, store head slice
#pragma unroll
    for (int r = 0; r < 8; ++r) {
        int m = r + 8 * hi;
        float s = __shfl(ssum, m, 32);
        attnOut[(bc + qbase + m) * Dd + head * 16 + row] = oAcc[r] / s;
    }
}

// ---------------------------------------------------------------------------
// out-proj + residual + LayerNorm1 (in-place on h). one block per token.
// ---------------------------------------------------------------------------
__global__ void projln_kernel(const float* __restrict__ attn,
                              const float* __restrict__ ow, const float* __restrict__ ob,
                              const float* __restrict__ g,  const float* __restrict__ bb,
                              float* __restrict__ h) {
    __shared__ float sA[64], sR[64];
    const int tok = blockIdx.x, t = threadIdx.x;
    sA[t] = attn[(size_t)tok * Dd + t];
    __syncthreads();
    float acc = ob[t];
    for (int dd = 0; dd < 64; ++dd) acc += sA[dd] * ow[t * 64 + dd];
    float r = h[(size_t)tok * Dd + t] + acc;
    sR[t] = r;
    __syncthreads();
    float m = 0.f;
    for (int dd = 0; dd < 64; ++dd) m += sR[dd];
    m *= (1.0f / 64.0f);
    float var = 0.f;
    for (int dd = 0; dd < 64; ++dd) { float d2 = sR[dd] - m; var += d2 * d2; }
    var *= (1.0f / 64.0f);
    h[(size_t)tok * Dd + t] = (r - m) * rsqrtf(var + 1e-5f) * g[t] + bb[t];
}

// ---------------------------------------------------------------------------
// FFN (64->256 gelu ->64) + residual + LayerNorm2 (in-place). block=256/token.
// ---------------------------------------------------------------------------
__global__ void __launch_bounds__(256)
ffn_kernel(float* __restrict__ h,
           const float* __restrict__ w1f, const float* __restrict__ b1f,
           const float* __restrict__ w2f, const float* __restrict__ b2f,
           const float* __restrict__ g,   const float* __restrict__ bb) {
    __shared__ float sH[64], sU[256], sR[64];
    const int tok = blockIdx.x, t = threadIdx.x;
    if (t < 64) sH[t] = h[(size_t)tok * Dd + t];
    __syncthreads();
    float u = b1f[t];
    for (int dd = 0; dd < 64; ++dd) u += sH[dd] * w1f[t * 64 + dd];
    sU[t] = gelu_f(u);
    __syncthreads();
    float r = 0.f;
    if (t < 64) {
        float acc = b2f[t];
        for (int uu = 0; uu < 256; ++uu) acc += sU[uu] * w2f[t * 256 + uu];
        r = sH[t] + acc;
        sR[t] = r;
    }
    __syncthreads();
    if (t < 64) {
        float m = 0.f;
        for (int dd = 0; dd < 64; ++dd) m += sR[dd];
        m *= (1.0f / 64.0f);
        float var = 0.f;
        for (int dd = 0; dd < 64; ++dd) { float d2 = sR[dd] - m; var += d2 * d2; }
        var *= (1.0f / 64.0f);
        h[(size_t)tok * Dd + t] = (r - m) * rsqrtf(var + 1e-5f) * g[t] + bb[t];
    }
}

// ---------------------------------------------------------------------------
// Task-query readout: softmax-pooled h -> head. one block per batch, 256 thr.
// ---------------------------------------------------------------------------
__global__ void __launch_bounds__(256)
readout_kernel(const float* __restrict__ h, const float* __restrict__ tq,
               const float* __restrict__ hw, const float* __restrict__ hbias,
               float* __restrict__ out) {
    __shared__ float sS[Cc];
    __shared__ float sRed[256];
    __shared__ float sP[64];
    const int b = blockIdx.x, t = threadIdx.x;
    const float* hbase = h + (size_t)b * Cc * Dd;
    float lmax = -INFINITY;
    for (int c = t; c < Cc; c += 256) {
        float s = 0.f;
        const float* hr = hbase + (size_t)c * Dd;
        for (int dd = 0; dd < 64; ++dd) s += hr[dd] * tq[dd];
        s *= 0.125f;                                  // 1/sqrt(64)
        sS[c] = s;
        lmax = fmaxf(lmax, s);
    }
    sRed[t] = lmax; __syncthreads();
    for (int st = 128; st; st >>= 1) {
        if (t < st) sRed[t] = fmaxf(sRed[t], sRed[t + st]);
        __syncthreads();
    }
    float M = sRed[0]; __syncthreads();
    float lsum = 0.f;
    for (int c = t; c < Cc; c += 256) lsum += expf(sS[c] - M);
    sRed[t] = lsum; __syncthreads();
    for (int st = 128; st; st >>= 1) {
        if (t < st) sRed[t] += sRed[t + st];
        __syncthreads();
    }
    float invS = 1.0f / sRed[0]; __syncthreads();
    if (t < 64) {
        float p = 0.f;
        for (int c = 0; c < Cc; ++c)
            p += expf(sS[c] - M) * invS * hbase[(size_t)c * Dd + t];
        sP[t] = p;
    }
    __syncthreads();
    if (t < OUTn) {
        float o = hbias[t];
        for (int dd = 0; dd < 64; ++dd) o += sP[dd] * hw[t * 64 + dd];
        out[b * OUTn + t] = o;
    }
}

// ---------------------------------------------------------------------------
extern "C" void kernel_launch(void* const* d_in, const int* in_sizes, int n_in,
                              void* d_out, int out_size, void* d_ws, size_t ws_size,
                              hipStream_t stream) {
    const float* x     = (const float*)d_in[0];
    const float* role  = (const float*)d_in[1];
    const float* fw    = (const float*)d_in[2];
    const float* q_w   = (const float*)d_in[3];
    const float* q_b   = (const float*)d_in[4];
    const float* k_w   = (const float*)d_in[5];
    const float* k_b   = (const float*)d_in[6];
    const float* v_w   = (const float*)d_in[7];
    const float* v_b   = (const float*)d_in[8];
    const float* w1    = (const float*)d_in[9];
    const float* b1    = (const float*)d_in[10];
    const float* w2    = (const float*)d_in[11];
    const float* b2    = (const float*)d_in[12];
    const float* out_w = (const float*)d_in[13];
    const float* out_b = (const float*)d_in[14];
    const float* ln1g  = (const float*)d_in[15];
    const float* ln1b  = (const float*)d_in[16];
    const float* f1_w  = (const float*)d_in[17];
    const float* f1_b  = (const float*)d_in[18];
    const float* f2_w  = (const float*)d_in[19];
    const float* f2_b  = (const float*)d_in[20];
    const float* ln2g  = (const float*)d_in[21];
    const float* ln2b  = (const float*)d_in[22];
    const float* tq    = (const float*)d_in[23];
    const float* hw    = (const float*)d_in[24];
    const float* hbias = (const float*)d_in[25];

    // workspace layout (fp32): 7 buffers of B*C*D = 1M floats each (28 MB)
    const size_t N = (size_t)Bb * Cc * Dd;   // 1048576
    float* ws   = (float*)d_ws;
    float* h    = ws;
    float* q    = ws + 1 * N;
    float* k    = ws + 2 * N;
    float* v    = ws + 3 * N;
    float* qp   = ws + 4 * N;
    float* kp   = ws + 5 * N;
    float* attn = ws + 6 * N;

    bind_kernel<<<(int)(N / 256), 256, 0, stream>>>(x, role, fw, h);

    for (int l = 0; l < Ll; ++l) {
        qkv_kernel<<<Bb * Cc, 64, 0, stream>>>(
            h, q_w + l * 4096, q_b + l * 64,
            k_w + l * 4096, k_b + l * 64,
            v_w + l * 4096, v_b + l * 64,
            w1 + l * 768, q, k, v, qp, kp);

        dim3 ag(Cc / 64, Hh, Bb);
        attn_kernel<<<ag, 128, 0, stream>>>(
            q, qp, k, kp, v,
            w1 + l * 768, b1 + l * 16, w2 + l * 16, b2 + l, attn);

        projln_kernel<<<Bb * Cc, 64, 0, stream>>>(
            attn, out_w + l * 4096, out_b + l * 64,
            ln1g + l * 64, ln1b + l * 64, h);

        ffn_kernel<<<Bb * Cc, 256, 0, stream>>>(
            h, f1_w + l * 16384, f1_b + l * 256,
            f2_w + l * 16384, f2_b + l * 64,
            ln2g + l * 64, ln2b + l * 64);
    }

    readout_kernel<<<Bb, 256, 0, stream>>>(h, tq, hw, hbias, (float*)d_out);
}
